// TreeLSTM_4294967296282
// MI455X (gfx1250) — matrix operand order repaired and checked by reading, not compiled, not run on hardware
//
#include <hip/hip_runtime.h>
#include <cstdint>
#include <cstddef>

// ---------------- problem constants ----------------
#define NN    50000     // nodes
#define IND   128       // input dim
#define HD    256       // hidden dim
#define GD    1024      // gate width = 4*HD
#define MAXL  96        // upper bound on tree depth (random tree height ~30)
#define NC    10        // classes

// ---------------- CDNA5 WMMA types ----------------
typedef __attribute__((ext_vector_type(16))) __bf16 v16bf;
typedef __attribute__((ext_vector_type(8)))  __bf16 v8bf;
typedef __attribute__((ext_vector_type(8)))  float  v8f;

union V16U { v16bf v; v8bf h[2]; };

// ---------------- workspace layout ----------------
static constexpr size_t AlignUp(size_t x) { return (x + 255) & ~size_t(255); }
static constexpr size_t NHs = size_t(NN) * HD;

static constexpr size_t OFF_META  = 0;                                   // 16 int (0:dmax, 1:D)
static constexpr size_t OFF_HIST  = AlignUp(OFF_META  + 16   * 4);       // MAXL int (per-depth counts)
static constexpr size_t OFF_START = AlignUp(OFF_HIST  + MAXL * 4);       // MAXL int
static constexpr size_t OFF_CUR   = AlignUp(OFF_START + MAXL * 4);       // MAXL int
static constexpr size_t OFF_INT_END = AlignUp(OFF_CUR + MAXL * 4);
static constexpr size_t OFF_D0    = OFF_INT_END;                         // depth ping
static constexpr size_t OFF_D1    = AlignUp(OFF_D0   + size_t(NN)*4);    // depth pong (final)
static constexpr size_t OFF_P0    = AlignUp(OFF_D1   + size_t(NN)*4);
static constexpr size_t OFF_P1    = AlignUp(OFF_P0   + size_t(NN)*4);
static constexpr size_t OFF_ORD   = AlignUp(OFF_P1   + size_t(NN)*4);    // nodes grouped by depth
static constexpr size_t OFF_WIH1  = AlignUp(OFF_ORD  + size_t(NN)*4);    // bf16 GD*IND
static constexpr size_t OFF_WHH1  = AlignUp(OFF_WIH1 + size_t(GD)*IND*2);
static constexpr size_t OFF_WIH2  = AlignUp(OFF_WHH1 + size_t(GD)*HD*2);
static constexpr size_t OFF_WHH2  = AlignUp(OFF_WIH2 + size_t(GD)*HD*2);
static constexpr size_t OFF_BS1   = AlignUp(OFF_WHH2 + size_t(GD)*HD*2); // f32 GD (b_ih+b_hh)
static constexpr size_t OFF_BS2   = AlignUp(OFF_BS1  + size_t(GD)*4);
static constexpr size_t OFF_XB    = AlignUp(OFF_BS2  + size_t(GD)*4);    // bf16 NN*IND (x copy)
static constexpr size_t OFF_ACC   = AlignUp(OFF_XB   + size_t(NN)*IND*2);// ah1,ac1,ah2,ac2,c2root (f32)

// ---------------- small helpers ----------------
__device__ __forceinline__ float sigm(float x) { return 1.f / (1.f + __expf(-x)); }

__device__ __forceinline__ void atomic_add_f32(float* p, float v) {
  // hardware GLOBAL_ATOMIC_ADD_F32 (no-return form)
  unsafeAtomicAdd(p, v);
}

// ---------------- setup kernels ----------------
__global__ void zero_i32_kernel(int* p, int n) {
  for (int i = blockIdx.x * blockDim.x + threadIdx.x; i < n; i += gridDim.x * blockDim.x) p[i] = 0;
}
__global__ void zero_f32_kernel(float* p, size_t n) {
  for (size_t i = blockIdx.x * blockDim.x + threadIdx.x; i < n; i += size_t(gridDim.x) * blockDim.x) p[i] = 0.f;
}

// pointer-doubling depth: d[i] = hops from i to root; root = NN-1 is self-parented
__global__ void tree_init_kernel(const int* __restrict__ parent, int* dd, int* dp) {
  int i = blockIdx.x * blockDim.x + threadIdx.x;
  if (i >= NN) return;
  int p = parent[i];
  dp[i] = p;
  dd[i] = (p == i) ? 0 : 1;
}
__global__ void tree_double_kernel(const int* __restrict__ sd, const int* __restrict__ sp,
                                   int* __restrict__ dd, int* __restrict__ dp) {
  int i = blockIdx.x * blockDim.x + threadIdx.x;
  if (i >= NN) return;
  int p = sp[i];
  dd[i] = sd[i] + sd[p];
  dp[i] = sp[p];
}

__global__ void hist_kernel(const int* __restrict__ depth, char* ws) {
  int i = blockIdx.x * blockDim.x + threadIdx.x;
  if (i >= NN) return;
  int d = depth[i];
  if (d >= MAXL) d = MAXL - 1;
  atomicAdd(&((int*)(ws + OFF_HIST))[d], 1);
  atomicMax(&((int*)(ws + OFF_META))[0], d);
}

__global__ void scan_kernel(char* ws) {
  int* meta  = (int*)(ws + OFF_META);
  int* hist  = (int*)(ws + OFF_HIST);
  int* start = (int*)(ws + OFF_START);
  int dmax = meta[0];
  if (dmax >= MAXL) dmax = MAXL - 1;
  meta[1] = dmax + 1;                 // D = number of levels
  int s = 0;
  for (int d = dmax; d >= 0; --d) {   // deepest level first in `order`
    start[d] = s;
    s += hist[d];
  }
}

__global__ void scatter_kernel(const int* __restrict__ depth, char* ws) {
  int i = blockIdx.x * blockDim.x + threadIdx.x;
  if (i >= NN) return;
  int d = depth[i];
  if (d >= MAXL) d = MAXL - 1;
  int* cur = (int*)(ws + OFF_CUR);
  const int* start = (const int*)(ws + OFF_START);
  int* order = (int*)(ws + OFF_ORD);
  int pos = atomicAdd(&cur[d], 1);
  order[start[d] + pos] = i;
}

__global__ void cvt_bf16_kernel(const float* __restrict__ src, __bf16* __restrict__ dst, int n) {
  for (int i = blockIdx.x * blockDim.x + threadIdx.x; i < n; i += gridDim.x * blockDim.x)
    dst[i] = (__bf16)src[i];
}
__global__ void bias_sum_kernel(const float* __restrict__ a, const float* __restrict__ b,
                                float* __restrict__ o, int n) {
  int i = blockIdx.x * blockDim.x + threadIdx.x;
  if (i < n) o[i] = a[i] + b[i];
}

// ---------------- per-level TreeLSTM kernel (WMMA bf16) ----------------
// Block = 256 threads = 8 waves, one 16-node tile per block.
// Each wave owns 2 of the 16 H-column tiles and computes all 4 gates for them.
__global__ __launch_bounds__(256) void level_kernel(
    int l, const int* __restrict__ parent, char* __restrict__ ws)
{
  const int* meta = (const int*)(ws + OFF_META);
  const int D = meta[1];
  if (l >= D) return;
  const int d = D - 1 - l;                       // process deepest level first
  const int cnt  = ((const int*)(ws + OFF_HIST))[d];
  const int tile0 = blockIdx.x * 16;
  if (tile0 >= cnt) return;
  const int base = ((const int*)(ws + OFF_START))[d];
  const int* order = (const int*)(ws + OFF_ORD);

  const __bf16* xb   = (const __bf16*)(ws + OFF_XB);
  const __bf16* Wih1 = (const __bf16*)(ws + OFF_WIH1);
  const __bf16* Whh1 = (const __bf16*)(ws + OFF_WHH1);
  const __bf16* Wih2 = (const __bf16*)(ws + OFF_WIH2);
  const __bf16* Whh2 = (const __bf16*)(ws + OFF_WHH2);
  const float*  bs1  = (const float*)(ws + OFF_BS1);
  const float*  bs2  = (const float*)(ws + OFF_BS2);
  float* ah1 = (float*)(ws + OFF_ACC);
  float* ac1 = ah1 + NHs;
  float* ah2 = ac1 + NHs;
  float* ac2 = ah2 + NHs;
  float* c2r = ac2 + NHs;

  __shared__ __align__(16) __bf16 Ax  [16][IND];   // x rows (bf16, async DMA) 4 KB
  __shared__ __align__(16) __bf16 Ah1s[16][HD];    // h1 state rows            8 KB
  __shared__ __align__(16) __bf16 Ah2s[16][HD];    // h2 state rows            8 KB
  __shared__ __align__(16) __bf16 Ac1s[16][HD];    // c1 output -> LSTM2 A     8 KB
  __shared__ int nodes[16];
  __shared__ int vld[16];

  const int t = threadIdx.x;
  if (t < 16) {
    int r = tile0 + t;
    int ok = (r < cnt);
    nodes[t] = ok ? order[base + r] : 0;
    vld[t] = ok;
  }
  __syncthreads();

  // ---- async-copy the 16x128 bf16 x-tile straight into LDS (ASYNCcnt path) ----
  // 256 threads x 16B = full 4KB tile in one GLOBAL_LOAD_ASYNC_TO_LDS_B128.
  // Padded rows fetch row 0: harmless, their outputs are masked before the atomics.
  {
    int m  = t >> 4;                     // row 0..15
    int kb = (t & 15) * 16;              // byte offset within 256B row
    unsigned voff = (unsigned)nodes[m] * (IND * 2) + (unsigned)kb;        // byte offset vs xb
    unsigned ldsa = (unsigned)(uintptr_t)(&Ax[0][0]) + (unsigned)(m * IND * 2 + kb);
    asm volatile("global_load_async_to_lds_b128 %0, %1, %2"
                 :: "v"(ldsa), "v"(voff), "s"(xb)
                 : "memory");
  }

  // Gather + f32->bf16 convert the h-state tiles (overlaps with the async x copy)
  for (int e = t; e < 16 * HD; e += 256) {
    int m = e >> 8, k = e & (HD - 1);
    size_t off = size_t(nodes[m]) * HD + k;
    float v1 = vld[m] ? ah1[off] : 0.f;
    float v2 = vld[m] ? ah2[off] : 0.f;
    Ah1s[m][k] = (__bf16)v1;
    Ah2s[m][k] = (__bf16)v2;
  }
  asm volatile("s_wait_asynccnt 0" ::: "memory");
  __syncthreads();

  const int wave = t >> 5;
  const int lane = t & 31;
  const int ln   = lane & 15;   // A: row M; B: column N; C: column N
  const int kh   = lane >> 4;   // K-half select per ISA layouts
  const int ht0  = wave * 2;    // first of this wave's two H-tiles

  v8f zero8 = 0.f;
  v8f acc[2][4];
  #pragma unroll
  for (int tt = 0; tt < 2; ++tt)
    #pragma unroll
    for (int g = 0; g < 4; ++g) acc[tt][g] = zero8;

  // ---- LSTM1 gates: x @ W_ih1^T  (K = 128) ----
  for (int kc = 0; kc < IND / 32; ++kc) {
    V16U a;
    const v8bf* pa = (const v8bf*)(&Ax[0][0] + ln * IND + kc * 32 + kh * 16);
    a.h[0] = pa[0]; a.h[1] = pa[1];                     // K {0..15} / {16..31}
    #pragma unroll
    for (int tt = 0; tt < 2; ++tt) {
      #pragma unroll
      for (int g = 0; g < 4; ++g) {
        int col = g * HD + (ht0 + tt) * 16 + ln;        // gate column
        v16bf b = *(const v16bf*)(Wih1 + size_t(col) * IND + kc * 32 + kh * 16);
        acc[tt][g] = __builtin_amdgcn_wmma_f32_16x16x32_bf16(
            false, a.v, false, b, (short)0, acc[tt][g], false, false);
      }
    }
  }
  // ---- LSTM1 gates: h @ W_hh1^T  (K = 256) ----
  for (int kc = 0; kc < HD / 32; ++kc) {
    V16U a;
    const v8bf* pa = (const v8bf*)(&Ah1s[0][0] + ln * HD + kc * 32 + kh * 16);
    a.h[0] = pa[0]; a.h[1] = pa[1];
    #pragma unroll
    for (int tt = 0; tt < 2; ++tt) {
      #pragma unroll
      for (int g = 0; g < 4; ++g) {
        int col = g * HD + (ht0 + tt) * 16 + ln;
        v16bf b = *(const v16bf*)(Whh1 + size_t(col) * HD + kc * 32 + kh * 16);
        acc[tt][g] = __builtin_amdgcn_wmma_f32_16x16x32_bf16(
            false, a.v, false, b, (short)0, acc[tt][g], false, false);
      }
    }
  }

  // ---- LSTM1 elementwise: C/D layout row m = j + kh*8, col n = tile*16 + ln ----
  #pragma unroll
  for (int tt = 0; tt < 2; ++tt) {
    int n = (ht0 + tt) * 16 + ln;
    float bi = bs1[n], bf = bs1[HD + n], bg = bs1[2 * HD + n], bo = bs1[3 * HD + n];
    #pragma unroll
    for (int j = 0; j < 8; ++j) {
      int m = j + kh * 8;
      int nd = nodes[m];
      int ok = vld[m];
      float cp = ok ? ac1[size_t(nd) * HD + n] : 0.f;
      float iv = sigm(acc[tt][0][j] + bi);
      float fv = sigm(acc[tt][1][j] + bf);
      float gv = tanhf(acc[tt][2][j] + bg);
      float ov = sigm(acc[tt][3][j] + bo);
      float c1 = fv * cp + iv * gv;
      float h1 = ov * tanhf(c1);
      Ac1s[m][n] = (__bf16)c1;                 // feeds LSTM2 as A-matrix
      if (ok) {
        int p = parent[nd];
        atomic_add_f32(&ah1[size_t(p) * HD + n], h1);
        atomic_add_f32(&ac1[size_t(p) * HD + n], c1);
      }
    }
  }
  __syncthreads();

  #pragma unroll
  for (int tt = 0; tt < 2; ++tt)
    #pragma unroll
    for (int g = 0; g < 4; ++g) acc[tt][g] = zero8;

  // ---- LSTM2 gates: c1 @ W_ih2^T  (K = 256) ----
  for (int kc = 0; kc < HD / 32; ++kc) {
    V16U a;
    const v8bf* pa = (const v8bf*)(&Ac1s[0][0] + ln * HD + kc * 32 + kh * 16);
    a.h[0] = pa[0]; a.h[1] = pa[1];
    #pragma unroll
    for (int tt = 0; tt < 2; ++tt) {
      #pragma unroll
      for (int g = 0; g < 4; ++g) {
        int col = g * HD + (ht0 + tt) * 16 + ln;
        v16bf b = *(const v16bf*)(Wih2 + size_t(col) * HD + kc * 32 + kh * 16);
        acc[tt][g] = __builtin_amdgcn_wmma_f32_16x16x32_bf16(
            false, a.v, false, b, (short)0, acc[tt][g], false, false);
      }
    }
  }
  // ---- LSTM2 gates: h2 @ W_hh2^T  (K = 256) ----
  for (int kc = 0; kc < HD / 32; ++kc) {
    V16U a;
    const v8bf* pa = (const v8bf*)(&Ah2s[0][0] + ln * HD + kc * 32 + kh * 16);
    a.h[0] = pa[0]; a.h[1] = pa[1];
    #pragma unroll
    for (int tt = 0; tt < 2; ++tt) {
      #pragma unroll
      for (int g = 0; g < 4; ++g) {
        int col = g * HD + (ht0 + tt) * 16 + ln;
        v16bf b = *(const v16bf*)(Whh2 + size_t(col) * HD + kc * 32 + kh * 16);
        acc[tt][g] = __builtin_amdgcn_wmma_f32_16x16x32_bf16(
            false, a.v, false, b, (short)0, acc[tt][g], false, false);
      }
    }
  }

  // ---- LSTM2 elementwise + scatter-add to parents + root capture ----
  #pragma unroll
  for (int tt = 0; tt < 2; ++tt) {
    int n = (ht0 + tt) * 16 + ln;
    float bi = bs2[n], bf = bs2[HD + n], bg = bs2[2 * HD + n], bo = bs2[3 * HD + n];
    #pragma unroll
    for (int j = 0; j < 8; ++j) {
      int m = j + kh * 8;
      int nd = nodes[m];
      int ok = vld[m];
      float cp = ok ? ac2[size_t(nd) * HD + n] : 0.f;
      float iv = sigm(acc[tt][0][j] + bi);
      float fv = sigm(acc[tt][1][j] + bf);
      float gv = tanhf(acc[tt][2][j] + bg);
      float ov = sigm(acc[tt][3][j] + bo);
      float c2 = fv * cp + iv * gv;
      float h2 = ov * tanhf(c2);
      if (ok) {
        int p = parent[nd];
        atomic_add_f32(&ah2[size_t(p) * HD + n], h2);
        atomic_add_f32(&ac2[size_t(p) * HD + n], c2);
        if (nd == NN - 1) c2r[n] = c2;          // root's c2 row
      }
    }
  }
}

// ---------------- final FC: out = c2root @ W_fc^T + b_fc ----------------
__global__ void fc_kernel(const float* __restrict__ Wfc, const float* __restrict__ bfc,
                          char* __restrict__ ws, float* __restrict__ out) {
  const float* c2r = (const float*)(ws + OFF_ACC) + 4 * NHs;
  int j = blockIdx.x;                 // one wave per class
  float s = 0.f;
  for (int k = threadIdx.x; k < HD; k += 32) s += c2r[k] * Wfc[j * HD + k];
  #pragma unroll
  for (int o = 16; o > 0; o >>= 1) s += __shfl_down(s, o, 32);
  if (threadIdx.x == 0) out[j] = s + bfc[j];
}

// ---------------- host side ----------------
extern "C" void kernel_launch(void* const* d_in, const int* in_sizes, int n_in,
                              void* d_out, int out_size, void* d_ws, size_t ws_size,
                              hipStream_t stream) {
  (void)in_sizes; (void)n_in; (void)out_size; (void)ws_size;
  const float* x      = (const float*)d_in[0];
  // d_in[1] = levels, d_in[2] = mask: rebuilt on-device from parent instead
  const int*   parent = (const int*)d_in[3];
  const float* Wih1   = (const float*)d_in[4];
  const float* Whh1   = (const float*)d_in[5];
  const float* bih1   = (const float*)d_in[6];
  const float* bhh1   = (const float*)d_in[7];
  const float* Wih2   = (const float*)d_in[8];
  const float* Whh2   = (const float*)d_in[9];
  const float* bih2   = (const float*)d_in[10];
  const float* bhh2   = (const float*)d_in[11];
  const float* Wfc    = (const float*)d_in[12];
  const float* bfc    = (const float*)d_in[13];
  char*  ws  = (char*)d_ws;
  float* out = (float*)d_out;

  const int g256 = (NN + 255) / 256;

  // 1) zero control block + state accumulators (+ c2root)
  zero_i32_kernel<<<64, 256, 0, stream>>>((int*)(ws + OFF_META), (int)(OFF_INT_END / 4));
  zero_f32_kernel<<<8192, 256, 0, stream>>>((float*)(ws + OFF_ACC), 4 * NHs + HD);

  // 2) depth via pointer doubling (9 passes -> covers depth up to 512)
  int* d0 = (int*)(ws + OFF_D0); int* d1 = (int*)(ws + OFF_D1);
  int* p0 = (int*)(ws + OFF_P0); int* p1 = (int*)(ws + OFF_P1);
  tree_init_kernel<<<g256, 256, 0, stream>>>(parent, d0, p0);
  for (int it = 0; it < 9; ++it) {
    if ((it & 1) == 0) tree_double_kernel<<<g256, 256, 0, stream>>>(d0, p0, d1, p1);
    else               tree_double_kernel<<<g256, 256, 0, stream>>>(d1, p1, d0, p0);
  }
  const int* depth = d1;  // final after odd (9) passes

  // 3) level histogram, offsets, counting-sort by depth
  hist_kernel<<<g256, 256, 0, stream>>>(depth, ws);
  scan_kernel<<<1, 1, 0, stream>>>(ws);
  scatter_kernel<<<g256, 256, 0, stream>>>(depth, ws);

  // 4) weights + x -> bf16 (weights L2-resident, ~1.8 MB), biases summed
  cvt_bf16_kernel<<<512, 256, 0, stream>>>(Wih1, (__bf16*)(ws + OFF_WIH1), GD * IND);
  cvt_bf16_kernel<<<1024, 256, 0, stream>>>(Whh1, (__bf16*)(ws + OFF_WHH1), GD * HD);
  cvt_bf16_kernel<<<1024, 256, 0, stream>>>(Wih2, (__bf16*)(ws + OFF_WIH2), GD * HD);
  cvt_bf16_kernel<<<1024, 256, 0, stream>>>(Whh2, (__bf16*)(ws + OFF_WHH2), GD * HD);
  cvt_bf16_kernel<<<8192, 256, 0, stream>>>(x,    (__bf16*)(ws + OFF_XB),   NN * IND);
  bias_sum_kernel<<<4, 256, 0, stream>>>(bih1, bhh1, (float*)(ws + OFF_BS1), GD);
  bias_sum_kernel<<<4, 256, 0, stream>>>(bih2, bhh2, (float*)(ws + OFF_BS2), GD);

  // 5) level-synchronous TreeLSTM: fixed MAXL launches, early-exit past D
  const int tiles = (NN + 15) / 16;  // worst-case level size
  for (int l = 0; l < MAXL; ++l)
    level_kernel<<<tiles, 256, 0, stream>>>(l, parent, ws);

  // 6) final 1x10 FC on root c2
  fc_kernel<<<NC, 32, 0, stream>>>(Wfc, bfc, ws, out);
}
